// WindowMixer_7370163880587
// MI455X (gfx1250) — compile-verified, matrix operand order checked by reading, not run on hardware
//
#include <hip/hip_runtime.h>

// ---------------------------------------------------------------------------
// Types
// ---------------------------------------------------------------------------
typedef __attribute__((ext_vector_type(16))) __bf16        bf16x16;
typedef __attribute__((ext_vector_type(8)))  float         f32x8;
typedef __attribute__((ext_vector_type(4)))  unsigned int  u32x4;

union Frag {
    bf16x16 v;
    u32x4   q[2];
    unsigned short h[16];
};

__device__ __forceinline__ unsigned short f2bf(float f) {
    unsigned int u = __float_as_uint(f);
    unsigned int r = u + 0x7FFFu + ((u >> 16) & 1u);   // round-to-nearest-even
    return (unsigned short)(r >> 16);
}

__device__ __forceinline__ f32x8 wmma_bf16(const Frag& a, const Frag& b, f32x8 c) {
    return __builtin_amdgcn_wmma_f32_16x16x32_bf16(
        /*neg_a=*/false, a.v, /*neg_b=*/false, b.v,
        /*c_mod=*/(short)0, c, /*reuse_a=*/false, /*reuse_b=*/false);
}

// ---------------------------------------------------------------------------
// Weight packer: fp32 row-major [K][N] -> bf16 B-fragment blobs.
// Blob = (ntile, ktile) 32x16 block; per lane 16 bf16 contiguous (32 bytes):
//   lane L: column N = n0 + (L&15);
//   K values: lanes 0-15 hold {k0..k0+7, k0+16..k0+23},
//             lanes 16-31 hold {k0+8..k0+15, k0+24..k0+31}
// (mirror of the documented 16-bit A-matrix layout).
// w_qkv: 72 ntiles x 12 ktiles = 864 blobs; w_out: 24 x 12 = 288 blobs.
// ---------------------------------------------------------------------------
__global__ void pack_weights(const float* __restrict__ w_qkv,
                             const float* __restrict__ w_out,
                             unsigned short* __restrict__ pk) {
    int tid = blockIdx.x * 256 + threadIdx.x;
    if (tid >= (864 + 288) * 32) return;
    int blob = tid >> 5;
    int lane = tid & 31;

    const float* src;
    int N, nt, ks;
    if (blob < 864) { src = w_qkv; N = 1152; nt = blob / 12;         ks = blob % 12; }
    else            { src = w_out; N = 384;  nt = (blob - 864) / 12; ks = (blob - 864) % 12; }

    int n     = nt * 16 + (lane & 15);
    int kbase = ks * 32 + ((lane & 16) ? 8 : 0);

    Frag t;
    #pragma unroll
    for (int j = 0; j < 16; ++j) {
        int v  = j >> 1;
        int kk = (v < 4 ? 2 * v : 16 + 2 * (v - 4)) + (j & 1);
        t.h[j] = f2bf(src[(size_t)(kbase + kk) * N + n]);
    }
    u32x4* o = (u32x4*)(pk + (size_t)blob * 512 + (size_t)lane * 16);
    o[0] = t.q[0];
    o[1] = t.q[1];
}

// ---------------------------------------------------------------------------
// Fused windowed attention: one workgroup per window (512 blocks, 256 thr).
// ---------------------------------------------------------------------------
__global__ __launch_bounds__(256, 1)
void window_attn(const float* __restrict__ x,
                 const float* __restrict__ b_qkv,
                 const float* __restrict__ b_out,
                 const unsigned short* __restrict__ pk,
                 float* __restrict__ out) {
    __shared__ unsigned short qbuf[64 * 384];   // q row-major [tok][384]
    __shared__ unsigned short kbuf[64 * 384];   // k row-major [tok][384]
    __shared__ unsigned short vtbuf[384 * 64];  // v transposed [ch][tok]
    __shared__ unsigned short xbuf[64 * 384];   // x window (bf16); reused as attn out
    __shared__ unsigned short pscr[8 * 16 * 64];// per-wave softmax P staging

    const int tid  = threadIdx.x;
    const int lane = tid & 31;
    const int wv   = tid >> 5;              // wave 0..7
    const int ln16 = lane & 15;
    const int grp8 = (lane & 16) ? 8 : 0;   // upper half-wave handles +8 (rows/K)

    const int b    = blockIdx.x;            // window id
    const int nimg = b >> 6;
    const int hh   = (b >> 3) & 7;
    const int wwk  = b & 7;

    // ---------------- Phase 1: stage window into LDS as bf16 ----------------
    {
        int row = tid >> 2;                 // 0..63 (token in window)
        int c0  = (tid & 3) * 96;
        int hw  = row >> 3, ww2 = row & 7;
        size_t g = ((size_t)nimg * 4096 + (hh * 8 + hw) * 64 + (wwk * 8 + ww2)) * 384;
        const float4* src = (const float4*)(x + g + c0);
        unsigned short* dst = &xbuf[row * 384 + c0];
        #pragma unroll 4
        for (int i = 0; i < 24; ++i) {
            float4 f = src[i];
            unsigned int lo = (unsigned int)f2bf(f.x) | ((unsigned int)f2bf(f.y) << 16);
            unsigned int hi = (unsigned int)f2bf(f.z) | ((unsigned int)f2bf(f.w) << 16);
            *(uint2*)(dst + i * 4) = make_uint2(lo, hi);
        }
    }
    __syncthreads();

    // ---------------- Phase 2: QKV GEMM  [64x384] @ [384x1152] --------------
    // N-tiles split as region(q|k|v) x 24 local tiles; wave parity interleave.
    {
        const int mt = wv >> 1;
        const int m0 = mt * 16;
        Frag A[12];
        const unsigned short* arow = &xbuf[(m0 + ln16) * 384];
        #pragma unroll
        for (int ks = 0; ks < 12; ++ks) {
            A[ks].q[0] = *(const u32x4*)(arow + ks * 32 + grp8);
            A[ks].q[1] = *(const u32x4*)(arow + ks * 32 + grp8 + 16);
        }
        #pragma unroll
        for (int region = 0; region < 3; ++region) {     // 0=q 1=k 2=v
            for (int jn = 0; jn < 12; ++jn) {
                int nl = (wv & 1) + 2 * jn;              // local ntile 0..23
                int nt = region * 24 + nl;
                float bias = b_qkv[nt * 16 + ln16];
                f32x8 acc = {bias, bias, bias, bias, bias, bias, bias, bias};
                const unsigned short* bp = pk + (size_t)(nt * 12) * 512 + (size_t)lane * 16;
                #pragma unroll
                for (int ks = 0; ks < 12; ++ks) {
                    Frag B;
                    B.q[0] = *(const u32x4*)(bp + ks * 512);
                    B.q[1] = *(const u32x4*)(bp + ks * 512 + 8);
                    acc = wmma_bf16(A[ks], B, acc);
                }
                int col = nl * 16 + ln16;
                if (region == 0) {
                    #pragma unroll
                    for (int r = 0; r < 8; ++r) qbuf[(m0 + r + grp8) * 384 + col] = f2bf(acc[r]);
                } else if (region == 1) {
                    #pragma unroll
                    for (int r = 0; r < 8; ++r) kbuf[(m0 + r + grp8) * 384 + col] = f2bf(acc[r]);
                } else {
                    #pragma unroll
                    for (int r = 0; r < 8; ++r) vtbuf[col * 64 + (m0 + r + grp8)] = f2bf(acc[r]);
                }
            }
        }
    }
    __syncthreads();

    // ---------------- Phase 3: attention (6 heads x 4 q-tiles) --------------
    {
        unsigned short* pb = &pscr[wv * 1024];
        #pragma unroll 1
        for (int i = 0; i < 3; ++i) {
            int u = wv * 3 + i;             // (head, qtile) unit
            int head = u >> 2;
            int qt   = u & 3;
            int hbase = head * 64;

            Frag QA[2];
            const unsigned short* qrow = &qbuf[(qt * 16 + ln16) * 384 + hbase];
            #pragma unroll
            for (int ks = 0; ks < 2; ++ks) {
                QA[ks].q[0] = *(const u32x4*)(qrow + ks * 32 + grp8);
                QA[ks].q[1] = *(const u32x4*)(qrow + ks * 32 + grp8 + 16);
            }

            // scores S[16x64] = Q @ K^T
            f32x8 s[4];
            #pragma unroll
            for (int t = 0; t < 4; ++t) {
                f32x8 acc = {0.f, 0.f, 0.f, 0.f, 0.f, 0.f, 0.f, 0.f};
                const unsigned short* krow = &kbuf[(t * 16 + ln16) * 384 + hbase];
                #pragma unroll
                for (int ks = 0; ks < 2; ++ks) {
                    Frag B;
                    B.q[0] = *(const u32x4*)(krow + ks * 32 + grp8);
                    B.q[1] = *(const u32x4*)(krow + ks * 32 + grp8 + 16);
                    acc = wmma_bf16(QA[ks], B, acc);
                }
                s[t] = acc;
            }

            // softmax per row (row M = r + 8*(lane/16); 16 cols across lanes x 4 tiles)
            #pragma unroll
            for (int r = 0; r < 8; ++r) {
                float v0 = s[0][r] * 0.125f, v1 = s[1][r] * 0.125f;
                float v2 = s[2][r] * 0.125f, v3 = s[3][r] * 0.125f;
                float m = fmaxf(fmaxf(v0, v1), fmaxf(v2, v3));
                #pragma unroll
                for (int msk = 8; msk >= 1; msk >>= 1) m = fmaxf(m, __shfl_xor(m, msk, 32));
                float e0 = __expf(v0 - m), e1 = __expf(v1 - m);
                float e2 = __expf(v2 - m), e3 = __expf(v3 - m);
                float sm = e0 + e1 + e2 + e3;
                #pragma unroll
                for (int msk = 8; msk >= 1; msk >>= 1) sm += __shfl_xor(sm, msk, 32);
                float inv = 1.0f / sm;
                int prow = (r + grp8) * 64 + ln16;
                pb[prow]      = f2bf(e0 * inv);
                pb[prow + 16] = f2bf(e1 * inv);
                pb[prow + 32] = f2bf(e2 * inv);
                pb[prow + 48] = f2bf(e3 * inv);
            }

            // out_h[16x64] = P @ V   (V fragments contiguous thanks to vT layout)
            Frag PA[2];
            const unsigned short* prow2 = &pb[ln16 * 64];
            #pragma unroll
            for (int ks = 0; ks < 2; ++ks) {
                PA[ks].q[0] = *(const u32x4*)(prow2 + ks * 32 + grp8);
                PA[ks].q[1] = *(const u32x4*)(prow2 + ks * 32 + grp8 + 16);
            }
            #pragma unroll
            for (int t = 0; t < 4; ++t) {
                f32x8 acc = {0.f, 0.f, 0.f, 0.f, 0.f, 0.f, 0.f, 0.f};
                int col = hbase + t * 16 + ln16;
                const unsigned short* vrow = &vtbuf[col * 64];
                #pragma unroll
                for (int ks = 0; ks < 2; ++ks) {
                    Frag B;
                    B.q[0] = *(const u32x4*)(vrow + ks * 32 + grp8);
                    B.q[1] = *(const u32x4*)(vrow + ks * 32 + grp8 + 16);
                    acc = wmma_bf16(PA[ks], B, acc);
                }
                #pragma unroll
                for (int r = 0; r < 8; ++r)
                    xbuf[(qt * 16 + r + grp8) * 384 + col] = f2bf(acc[r]);
            }
        }
    }
    __syncthreads();

    // ---------------- Phase 4: out projection [64x384] @ [384x384] ----------
    {
        const int mt = wv >> 1;
        const int m0 = mt * 16;
        Frag A[12];
        const unsigned short* arow = &xbuf[(m0 + ln16) * 384];
        #pragma unroll
        for (int ks = 0; ks < 12; ++ks) {
            A[ks].q[0] = *(const u32x4*)(arow + ks * 32 + grp8);
            A[ks].q[1] = *(const u32x4*)(arow + ks * 32 + grp8 + 16);
        }
        // Hoist un-window output row addresses (fixed per wave across N-tiles).
        size_t orow[8];
        #pragma unroll
        for (int r = 0; r < 8; ++r) {
            int tok = m0 + r + grp8;
            int hw = tok >> 3, ww2 = tok & 7;
            orow[r] = ((size_t)nimg * 4096 + (hh * 8 + hw) * 64 + (wwk * 8 + ww2)) * 384;
        }
        for (int jn = 0; jn < 12; ++jn) {
            int nt = (wv & 1) + 2 * jn;
            float bias = b_out[nt * 16 + ln16];
            f32x8 acc = {bias, bias, bias, bias, bias, bias, bias, bias};
            const unsigned short* bp = pk + (size_t)(864 + nt * 12) * 512 + (size_t)lane * 16;
            #pragma unroll
            for (int ks = 0; ks < 12; ++ks) {
                Frag B;
                B.q[0] = *(const u32x4*)(bp + ks * 512);
                B.q[1] = *(const u32x4*)(bp + ks * 512 + 8);
                acc = wmma_bf16(A[ks], B, acc);
            }
            int col = nt * 16 + ln16;
            #pragma unroll
            for (int r = 0; r < 8; ++r)
                out[orow[r] + col] = acc[r];
        }
    }
}

// ---------------------------------------------------------------------------
extern "C" void kernel_launch(void* const* d_in, const int* in_sizes, int n_in,
                              void* d_out, int out_size, void* d_ws, size_t ws_size,
                              hipStream_t stream) {
    const float* x     = (const float*)d_in[0];
    const float* w_qkv = (const float*)d_in[1];
    const float* b_qkv = (const float*)d_in[2];
    const float* w_out = (const float*)d_in[3];
    const float* b_out = (const float*)d_in[4];
    float* out = (float*)d_out;
    unsigned short* pk = (unsigned short*)d_ws;   // needs (864+288)*512*2 = 1.18 MB

    pack_weights<<<144, 256, 0, stream>>>(w_qkv, w_out, pk);
    window_attn<<<512, 256, 0, stream>>>(x, b_qkv, b_out, pk, out);
}
